// MGCNN_11493332484278
// MI455X (gfx1250) — matrix operand order repaired and checked by reading, not compiled, not run on hardware
//
#include <hip/hip_runtime.h>
#include <hip/hip_bf16.h>
#include <math.h>

#define NNODE 100000
#define NEDGE 3200000
#define RANK  10
#define NCONV 32
#define ORD   5
#define NITER 10
#define NBATCH 65536
#define KFEAT (ORD * RANK)   // 50 real feature cols, padded to 64

typedef __bf16 bf16_t;
typedef __attribute__((ext_vector_type(16))) __bf16 v16bf;
typedef __attribute__((ext_vector_type(8)))  float  v8f;

__device__ __forceinline__ float sigmoid_(float x) { return 1.0f / (1.0f + __expf(-x)); }

// ---------------- utility kernels ----------------
__global__ void k_zero(float* __restrict__ p, int n) {
    int i = blockIdx.x * blockDim.x + threadIdx.x;
    if (i < n) p[i] = 0.0f;
}

__global__ void k_copy(float* __restrict__ d, const float* __restrict__ s, int n) {
    int i = blockIdx.x * blockDim.x + threadIdx.x;
    if (i < n) d[i] = s[i];
}

// deg[row] += val  (graph degree via HW f32 atomics; L2-resident accumulator)
__global__ void k_deg(const int* __restrict__ row, const float* __restrict__ val,
                      float* __restrict__ deg, int e) {
    int i = blockIdx.x * blockDim.x + threadIdx.x;
    if (i < e) unsafeAtomicAdd(&deg[row[i]], val[i]);
}

// deg -> deg^-0.5 in place
__global__ void k_dinv(float* __restrict__ d, int n) {
    int i = blockIdx.x * blockDim.x + threadIdx.x;
    if (i < n) { float v = d[i]; d[i] = (v > 0.0f) ? rsqrtf(v) : 0.0f; }
}

// feat[:, 0:RANK] = X   (Chebyshev T0)
__global__ void k_feat0(float* __restrict__ feat, const float* __restrict__ X, int n) {
    int i = blockIdx.x * blockDim.x + threadIdx.x;
    if (i < n * RANK) {
        int node = i / RANK, c = i % RANK;
        feat[(size_t)node * KFEAT + c] = X[i];
    }
}

// acc[row] += val * dinv[col] * feat[col, block k-1]   (one thread per edge)
__global__ void k_spmm(const int* __restrict__ row, const int* __restrict__ col,
                       const float* __restrict__ val, const float* __restrict__ dinv,
                       const float* __restrict__ feat, int koff,
                       float* __restrict__ acc, int e) {
    int i = blockIdx.x * blockDim.x + threadIdx.x;
    if (i >= e) return;
    int r = row[i], c = col[i];
    float s = val[i] * dinv[c];
    const float* y = feat + (size_t)c * KFEAT + koff;
    float* o = acc + (size_t)r * RANK;
#pragma unroll
    for (int j = 0; j < RANK; ++j) unsafeAtomicAdd(&o[j], s * y[j]);
}

// T_k = (k>=2 ? 2 : 1) * (T_{k-1} - dinv * acc) - (k>=2 ? T_{k-2} : 0); acc -> 0
__global__ void k_combine(float* __restrict__ feat, float* __restrict__ acc,
                          const float* __restrict__ dinv, int k, int n) {
    int i = blockIdx.x * blockDim.x + threadIdx.x;
    if (i >= n * RANK) return;
    int node = i / RANK, c = i % RANK;
    size_t base = (size_t)node * KFEAT;
    float y = feat[base + (k - 1) * RANK + c];
    float l = y - dinv[node] * acc[i];
    float t = (k >= 2) ? (2.0f * l - feat[base + (k - 2) * RANK + c]) : l;
    feat[base + k * RANK + c] = t;
    acc[i] = 0.0f;   // leave zeroed for next SpMM pass
}

// pack weights to bf16, padded for WMMA:
//   WcP  [64][32]  (rows 50..63 zero)
//   Wcat [64][128] (rows 0..31 = Wg[g][f][o], rows 32..63 = Ug; col = g*32+o)
//   WoP  [32][16]  (cols 10..15 zero)
__global__ void k_prep(const float* __restrict__ Wc, const float* __restrict__ Wg,
                       const float* __restrict__ Ug, const float* __restrict__ Wo,
                       bf16_t* __restrict__ WcP, bf16_t* __restrict__ Wcat,
                       bf16_t* __restrict__ WoP) {
    const int TOT = 64 * 32 + 64 * 128 + 32 * 16;
    for (int i = blockIdx.x * blockDim.x + threadIdx.x; i < TOT; i += gridDim.x * blockDim.x) {
        if (i < 64 * 32) {
            int k = i / 32, o = i % 32;
            WcP[i] = (bf16_t)((k < KFEAT) ? Wc[k * 32 + o] : 0.0f);
        } else if (i < 64 * 32 + 64 * 128) {
            int j = i - 64 * 32;
            int kk = j / 128, cc = j % 128;
            int g = cc / 32, o = cc % 32;
            float v = (kk < 32) ? Wg[g * 1024 + kk * 32 + o] : Ug[g * 1024 + (kk - 32) * 32 + o];
            Wcat[j] = (bf16_t)v;
        } else {
            int j = i - 64 * 32 - 64 * 128;
            int f = j / 16, o = j % 16;
            WoP[j] = (bf16_t)((o < RANK) ? Wo[f * RANK + o] : 0.0f);
        }
    }
}

// -------- fused dense pipeline: x=relu(feat@Wc+bc); LSTM; X+=tanh(h'@Wo+bo) --------
// One wave per 16-node tile; all matrix math on v_wmma_f32_16x16x32_bf16.
__global__ __launch_bounds__(256) void k_dense(
    const float* __restrict__ feat,
    float* __restrict__ X, float* __restrict__ h, float* __restrict__ c,
    const bf16_t* __restrict__ WcP, const float* __restrict__ bc,
    const bf16_t* __restrict__ Wcat, const float* __restrict__ bg,
    const bf16_t* __restrict__ WoP, const float* __restrict__ bo,
    int nTiles)
{
    __shared__ float ldsAll[8][16][64];     // 4 KB per wave, intra-wave only
    const int lane = threadIdx.x & 31;
    const int wave = threadIdx.x >> 5;
    const int tile = blockIdx.x * 8 + wave;
    if (tile >= nTiles) return;             // wave-uniform: EXEC stays all-ones
    float (*T)[64] = ldsAll[wave];
    const int row0 = tile * 16;

    const int m   = lane & 15;              // M (A,C/D rows) or N (B,C/D cols)
    const int kbA = (lane < 16) ? 0 : 16;   // 16-bit A frag K-base
    const int kbB = (lane < 16) ? 0 : 16;   // 16-bit B frag K-base
    const int mb  = (lane < 16) ? 0 : 8;    // f32 C/D M-base

    // ---- stage feat tile (16 x 50 -> 16 x 64, zero padded) ----
    for (int i = lane; i < 16 * 64; i += 32) {
        int r = i >> 6, k = i & 63;
        T[r][k] = (k < KFEAT) ? feat[(size_t)(row0 + r) * KFEAT + k] : 0.0f;
    }
    __builtin_amdgcn_wave_barrier();

    v16bf a0, a1;                           // A frags, K = 0..31 and 32..63
#pragma unroll
    for (int j = 0; j < 16; ++j) {
        int k = kbA + j;   // contiguous K order, matches B
        a0[j] = (bf16_t)T[m][k];
        a1[j] = (bf16_t)T[m][k + 32];
    }

    // ---- GEMM1: x = relu(feat @ Wc + bc), [16 x 32] ----
    v8f xacc[2];
#pragma unroll
    for (int t = 0; t < 2; ++t) {
        v8f acc = {};
#pragma unroll
        for (int s = 0; s < 2; ++s) {
            v16bf b;
#pragma unroll
            for (int j = 0; j < 16; ++j)
                b[j] = WcP[(size_t)(kbB + 32 * s + j) * 32 + (t * 16 + m)];
            acc = __builtin_amdgcn_wmma_f32_16x16x32_bf16(
                false, (s == 0 ? a0 : a1), false, b, (short)0, acc, false, false);
        }
        float bias = bc[t * 16 + m];
#pragma unroll
        for (int r = 0; r < 8; ++r) {
            float v = acc[r] + bias;
            acc[r] = v > 0.0f ? v : 0.0f;
        }
        xacc[t] = acc;
    }

    // ---- stage [x | h] tile ----
#pragma unroll
    for (int t = 0; t < 2; ++t)
#pragma unroll
        for (int r = 0; r < 8; ++r)
            T[mb + r][t * 16 + m] = xacc[t][r];
    for (int i = lane; i < 16 * 32; i += 32) {
        int r = i >> 5, k = i & 31;
        T[r][32 + k] = h[(size_t)(row0 + r) * NCONV + k];
    }
    __builtin_amdgcn_wave_barrier();

    v16bf ax0, ax1;
#pragma unroll
    for (int j = 0; j < 16; ++j) {
        int k = kbA + j;
        ax0[j] = (bf16_t)T[m][k];
        ax1[j] = (bf16_t)T[m][k + 32];
    }
    __builtin_amdgcn_wave_barrier();

    // ---- GEMM2 + LSTM state update, per 16-column half (keeps 4 gate tiles live) ----
#pragma unroll
    for (int tt = 0; tt < 2; ++tt) {
        v8f g[4];                            // f, i, o, u
#pragma unroll
        for (int gi = 0; gi < 4; ++gi) {
            int nt = 2 * gi + tt;            // N-tile index into 128 gate columns
            v8f acc = {};
#pragma unroll
            for (int s = 0; s < 2; ++s) {
                v16bf b;
#pragma unroll
                for (int j = 0; j < 16; ++j)
                    b[j] = Wcat[(size_t)(kbB + 32 * s + j) * 128 + (nt * 16 + m)];
                acc = __builtin_amdgcn_wmma_f32_16x16x32_bf16(
                    false, (s == 0 ? ax0 : ax1), false, b, (short)0, acc, false, false);
            }
            float bias = bg[gi * NCONV + tt * 16 + m];
#pragma unroll
            for (int r = 0; r < 8; ++r) acc[r] = sigmoid_(acc[r] + bias);
            g[gi] = acc;
        }
#pragma unroll
        for (int r = 0; r < 8; ++r) {
            int rr = mb + r;
            size_t idx = (size_t)(row0 + rr) * NCONV + tt * 16 + m;
            float cold = c[idx];
            float cn = g[0][r] * cold + g[1][r] * g[3][r];
            float hn = g[2][r] * sigmoid_(cn);       // faithful: h = o * sigmoid(c)
            c[idx] = cn;
            h[idx] = hn;
            T[rr][tt * 16 + m] = hn;                 // stage h' for GEMM3
        }
    }
    __builtin_amdgcn_wave_barrier();

    // ---- GEMM3: X += tanh(h' @ Wout + bo) ----
    v16bf ah, bw;
#pragma unroll
    for (int j = 0; j < 16; ++j) {
        int k = kbA + j;
        ah[j] = (bf16_t)T[m][k];
        bw[j] = WoP[(size_t)(kbB + j) * 16 + m];
    }
    v8f oacc = {};
    oacc = __builtin_amdgcn_wmma_f32_16x16x32_bf16(
        false, ah, false, bw, (short)0, oacc, false, false);
    if (m < RANK) {
        float bias = bo[m];
#pragma unroll
        for (int r = 0; r < 8; ++r) {
            size_t idx = (size_t)(row0 + mb + r) * RANK + m;
            X[idx] += tanhf(oacc[r] + bias);
        }
    }
}

// out[b] = R_MIN + R_RANGE * sigmoid(dot(W[uid], H[iid]))
__global__ void k_predict(const int* __restrict__ uid, const int* __restrict__ iid,
                          const float* __restrict__ W, const float* __restrict__ H,
                          float* __restrict__ out, int n) {
    int i = blockIdx.x * blockDim.x + threadIdx.x;
    if (i >= n) return;
    const float* w = W + (size_t)uid[i] * RANK;
    const float* hh = H + (size_t)iid[i] * RANK;
    float r = 0.0f;
#pragma unroll
    for (int j = 0; j < RANK; ++j) r += w[j] * hh[j];
    out[i] = 1.0f + 4.0f * sigmoid_(r);
}

// ---------------- host orchestration ----------------
extern "C" void kernel_launch(void* const* d_in, const int* in_sizes, int n_in,
                              void* d_out, int out_size, void* d_ws, size_t ws_size,
                              hipStream_t stream) {
    (void)in_sizes; (void)n_in; (void)out_size; (void)ws_size;
    const int*   user_id = (const int*)d_in[0];
    const int*   item_id = (const int*)d_in[1];
    const int*   ug_row  = (const int*)d_in[2];
    const int*   ug_col  = (const int*)d_in[3];
    const float* ug_val  = (const float*)d_in[4];
    const int*   ig_row  = (const int*)d_in[5];
    const int*   ig_col  = (const int*)d_in[6];
    const float* ig_val  = (const float*)d_in[7];
    const float* W0      = (const float*)d_in[8];
    const float* H0      = (const float*)d_in[9];
    const float* WcW     = (const float*)d_in[10];
    const float* bcW     = (const float*)d_in[11];
    const float* WcH     = (const float*)d_in[12];
    const float* bcH     = (const float*)d_in[13];
    const float* WgU     = (const float*)d_in[14];
    const float* UgU     = (const float*)d_in[15];
    const float* bgU     = (const float*)d_in[16];
    const float* WgM     = (const float*)d_in[17];
    const float* UgM     = (const float*)d_in[18];
    const float* bgM     = (const float*)d_in[19];
    const float* WoW     = (const float*)d_in[20];
    const float* boW     = (const float*)d_in[21];
    const float* WoH     = (const float*)d_in[22];
    const float* boH     = (const float*)d_in[23];

    const int N = NNODE, E = NEDGE;
    char* ws = (char*)d_ws;
    auto carve = [&](size_t bytes) -> char* {
        char* p = ws;
        ws += (bytes + 255) & ~(size_t)255;
        return p;
    };
    float* W     = (float*)carve((size_t)N * RANK  * 4);
    float* H     = (float*)carve((size_t)N * RANK  * 4);
    float* featU = (float*)carve((size_t)N * KFEAT * 4);
    float* featM = (float*)carve((size_t)N * KFEAT * 4);
    float* accU  = (float*)carve((size_t)N * RANK  * 4);
    float* accM  = (float*)carve((size_t)N * RANK  * 4);
    float* hU    = (float*)carve((size_t)N * NCONV * 4);
    float* cU    = (float*)carve((size_t)N * NCONV * 4);
    float* hM    = (float*)carve((size_t)N * NCONV * 4);
    float* cM    = (float*)carve((size_t)N * NCONV * 4);
    float* dinvU = (float*)carve((size_t)N * 4);
    float* dinvM = (float*)carve((size_t)N * 4);
    bf16_t* WcPU  = (bf16_t*)carve(64 * 32 * 2);
    bf16_t* WcPM  = (bf16_t*)carve(64 * 32 * 2);
    bf16_t* WcatU = (bf16_t*)carve(64 * 128 * 2);
    bf16_t* WcatM = (bf16_t*)carve(64 * 128 * 2);
    bf16_t* WoPU  = (bf16_t*)carve(32 * 16 * 2);
    bf16_t* WoPM  = (bf16_t*)carve(32 * 16 * 2);
    // total workspace ~105 MB

    const int TB = 256;
    auto nb = [](int n) { return (n + 255) / 256; };
    const int nTiles = N / 16;                       // 6250 exactly
    const int denseBlocks = (nTiles + 7) / 8;

    // init state (re-done every call: deterministic, workspace may be poisoned)
    k_copy<<<nb(N * RANK), TB, 0, stream>>>(W, W0, N * RANK);
    k_copy<<<nb(N * RANK), TB, 0, stream>>>(H, H0, N * RANK);
    k_zero<<<nb(N * NCONV), TB, 0, stream>>>(hU, N * NCONV);
    k_zero<<<nb(N * NCONV), TB, 0, stream>>>(cU, N * NCONV);
    k_zero<<<nb(N * NCONV), TB, 0, stream>>>(hM, N * NCONV);
    k_zero<<<nb(N * NCONV), TB, 0, stream>>>(cM, N * NCONV);
    k_zero<<<nb(N * RANK), TB, 0, stream>>>(accU, N * RANK);
    k_zero<<<nb(N * RANK), TB, 0, stream>>>(accM, N * RANK);
    k_zero<<<nb(N), TB, 0, stream>>>(dinvU, N);
    k_zero<<<nb(N), TB, 0, stream>>>(dinvM, N);
    k_prep<<<16, TB, 0, stream>>>(WcW, WgU, UgU, WoW, WcPU, WcatU, WoPU);
    k_prep<<<16, TB, 0, stream>>>(WcH, WgM, UgM, WoH, WcPM, WcatM, WoPM);
    k_deg<<<nb(E), TB, 0, stream>>>(ug_row, ug_val, dinvU, E);
    k_deg<<<nb(E), TB, 0, stream>>>(ig_row, ig_val, dinvM, E);
    k_dinv<<<nb(N), TB, 0, stream>>>(dinvU, N);
    k_dinv<<<nb(N), TB, 0, stream>>>(dinvM, N);

    for (int it = 0; it < NITER; ++it) {
        // user graph
        k_feat0<<<nb(N * RANK), TB, 0, stream>>>(featU, W, N);
        for (int k = 1; k < ORD; ++k) {
            k_spmm<<<nb(E), TB, 0, stream>>>(ug_row, ug_col, ug_val, dinvU,
                                             featU, (k - 1) * RANK, accU, E);
            k_combine<<<nb(N * RANK), TB, 0, stream>>>(featU, accU, dinvU, k, N);
        }
        k_dense<<<denseBlocks, TB, 0, stream>>>(featU, W, hU, cU,
                                                WcPU, bcW, WcatU, bgU, WoPU, boW, nTiles);
        // item graph
        k_feat0<<<nb(N * RANK), TB, 0, stream>>>(featM, H, N);
        for (int k = 1; k < ORD; ++k) {
            k_spmm<<<nb(E), TB, 0, stream>>>(ig_row, ig_col, ig_val, dinvM,
                                             featM, (k - 1) * RANK, accM, E);
            k_combine<<<nb(N * RANK), TB, 0, stream>>>(featM, accM, dinvM, k, N);
        }
        k_dense<<<denseBlocks, TB, 0, stream>>>(featM, H, hM, cM,
                                                WcPM, bcH, WcatM, bgM, WoPM, boH, nTiles);
    }

    k_predict<<<nb(NBATCH), TB, 0, stream>>>(user_id, item_id, W, H, (float*)d_out, NBATCH);
}
